// MainModelCrossAttention_31147102830914
// MI455X (gfx1250) — matrix operand order repaired and checked by reading, not compiled, not run on hardware
//
#include <hip/hip_runtime.h>
#include <hip/hip_bf16.h>

// ---------------------------------------------------------------------------
// MI455X (gfx1250) cross-attention, bf16 WMMA pipeline with f32 accumulate.
// Shapes (fixed by the reference): B=8, H=1, Sq=Sk=2048, HIDDEN=2048.
// GEMMs: block tile 256x128, wave tile 64x64 (4x4 WMMA accums), K-step 32.
// LDS double-buffered, staged with GLOBAL_LOAD_ASYNC_TO_LDS_B128
// (ASYNCcnt-tracked, ISA 08_async_tensor §4), one barrier per K-step.
// ---------------------------------------------------------------------------

typedef unsigned short u16;
typedef __attribute__((ext_vector_type(16))) __bf16 v16bf;
typedef __attribute__((ext_vector_type(8)))  float  v8f;

#define HID 2048
#define SEQ 2048
#define NB  8

// round-to-nearest-even f32 -> bf16 (bits)
__device__ __forceinline__ u16 f2bf(float x) {
  union { float f; unsigned u; } c; c.f = x;
  unsigned r = c.u + 0x7fffu + ((c.u >> 16) & 1u);
  return (u16)(r >> 16);
}

__global__ __launch_bounds__(256) void cvt_f32_bf16(const float* __restrict__ in,
                                                    u16* __restrict__ out,
                                                    long long n) {
  long long i = (long long)blockIdx.x * blockDim.x + threadIdx.x;
  long long stride = (long long)gridDim.x * blockDim.x;
  for (; i < n; i += stride) out[i] = f2bf(in[i]);
}

// Async copy 16 bytes global -> LDS (per lane). Tracked by ASYNCcnt.
__device__ __forceinline__ void async_b128(unsigned lds_byte_addr, const void* gptr) {
  asm volatile("global_load_async_to_lds_b128 %0, %1, off"
               :: "v"(lds_byte_addr), "v"(gptr) : "memory");
}
__device__ __forceinline__ void wait_async0() {
  asm volatile("s_wait_asynccnt 0x0" ::: "memory");
}

// --------------------------- WMMA fragment loads ---------------------------
// LDS tiles are [rows][K_STEP=32] bf16 padded to 40 halves/row (80B rows,
// 16B aligned, conflict-free 16B loads across a half-wave).
//
// A (16x32, 16-bit) lane layout (ISA 7.12.2): lanes 0-15 row M=l, elements
// 0..7 = K 0..7, 8..15 = K 16..23; lanes 16-31 row M=l-16, K 8..15 / 24..31.
// B (32x16): lanes 0-15 col N=l hold K 0..7 / 16..23; lanes 16-31 K 8..15 / 24..31.

union FragU { v16bf v; uint4 q[2]; };

__device__ __forceinline__ v16bf load_frag_a(const u16* As, int row0, int lane) {
  int row = row0 + (lane & 15);
  const u16* p = As + row * 40 + ((lane >> 4) & 1) * 8;
  FragU f;
  f.q[0] = *(const uint4*)(p);       // K = hi*8 + 0..7
  f.q[1] = *(const uint4*)(p + 16);  // K = hi*8 + 16..23
  return f.v;
}

__device__ __forceinline__ v16bf load_frag_b(const u16* Bs, int col0, int lane) {
  int col = col0 + (lane & 15);
  const u16* p = Bs + col * 40 + ((lane >> 4) & 1) * 8;
  FragU f;
  f.q[0] = *(const uint4*)(p);       // K = hi*8 + 0..7
  f.q[1] = *(const uint4*)(p + 16);  // K = hi*8 + 16..23
  return f.v;
}

// 16 ds_load_b128 + 16 WMMA on one staged K-step tile (64x64 wave tile).
__device__ __forceinline__ void do_tile(const u16* Asb, const u16* Bsb,
                                        int waveM, int waveN, int lane,
                                        v8f (&acc)[4][4]) {
  v16bf af[4], bf[4];
#pragma unroll
  for (int tm = 0; tm < 4; ++tm) af[tm] = load_frag_a(Asb, waveM * 64 + tm * 16, lane);
#pragma unroll
  for (int tn = 0; tn < 4; ++tn) bf[tn] = load_frag_b(Bsb, waveN * 64 + tn * 16, lane);
#pragma unroll
  for (int tm = 0; tm < 4; ++tm)
#pragma unroll
    for (int tn = 0; tn < 4; ++tn)
      acc[tm][tn] = __builtin_amdgcn_wmma_f32_16x16x32_bf16(
          false, af[tm], false, bf[tn], (short)0, acc[tm][tn], false, false);
}

// ------------------------------- GEMM kernel -------------------------------
// D[M,N] = A[M,K] * B^T  (+ epilogue), A row-major MxK bf16.
// BK_MODE 0: B row-major [N,K] (weights / keys) -> fully async staged
// BK_MODE 1: B row-major [K,N] (values)         -> transposed reg staging
// EPI 0: bf16 D = acc + bias[n]
// EPI 1: f32  D = acc * scale
// EPI 2: bf16 D = acc + residual_f32[m*ldd+n]   (per-batch residual)
// EPI 3: f32  D = acc + bias[n]
// Block tile 256x128, 8 waves (4 M x 2 N); wave tile 64x64; K-step 32.
// M multiple of 256, N multiple of 128, K multiple of 32 (true for all calls).
template<int BK_MODE, int EPI>
__global__ __launch_bounds__(256) void gemm_wmma(
    const u16* __restrict__ A, long long sAb,
    const u16* __restrict__ Bm, long long sBb,
    void* __restrict__ Dp, long long sDb,
    const float* __restrict__ aux, long long sXb,
    int M, int N, int K, int lda, int ldb, int ldd, float scale)
{
  __shared__ __align__(16) u16 As[2][256 * 40];  // 2 x 20480 B
  __shared__ __align__(16) u16 Bs[2][128 * 40];  // 2 x 10240 B
  const unsigned BUFA = 256 * 40 * 2;
  const unsigned BUFB = 128 * 40 * 2;

  const int t     = threadIdx.x;
  const int lane  = t & 31;
  const int wave  = t >> 5;
  const int waveM = wave & 3;   // 4 wave-rows of 64
  const int waveN = wave >> 2;  // 2 wave-cols of 64
  const int bM    = blockIdx.y * 256;
  const int bN    = blockIdx.x * 128;
  const int bz    = blockIdx.z;

  A  += (long long)bz * sAb;
  Bm += (long long)bz * sBb;
  const float* auxp = aux + (long long)bz * sXb;

  // Staging: thread t owns rows (t>>2)+64c, column group (t&3) -> 16B chunks.
  const int sr = t >> 2, scg = t & 3;
  long long aOff[4];
  unsigned  ldsA[4];
  const unsigned ldsAbase = (unsigned)(size_t)&As[0][sr * 40 + scg * 8];
#pragma unroll
  for (int c = 0; c < 4; ++c) {
    aOff[c] = (long long)(bM + sr + 64 * c) * lda + scg * 8;
    ldsA[c] = ldsAbase + (unsigned)(c * 64 * 40 * 2);
  }
  const long long bOff0 = (long long)(bN + sr) * ldb + scg * 8;  // BK_MODE 0
  const long long bOff1 = bOff0 + 64LL * ldb;
  const unsigned ldsB0 = (unsigned)(size_t)&Bs[0][sr * 40 + scg * 8];
  const unsigned ldsB1 = ldsB0 + 64 * 40 * 2;
  // BK_MODE 1 (transpose) staging: thread t owns k-row (t&31), 16 columns.
  const int kk = t & 31, ng = t >> 5;

  v8f zero = {0.f, 0.f, 0.f, 0.f, 0.f, 0.f, 0.f, 0.f};
  v8f acc[4][4];
#pragma unroll
  for (int i = 0; i < 4; ++i)
#pragma unroll
    for (int j = 0; j < 4; ++j) acc[i][j] = zero;

  const int nsteps = K >> 5;

  // ---- prologue: stage tile 0 into buffer 0
#pragma unroll
  for (int c = 0; c < 4; ++c) async_b128(ldsA[c], A + aOff[c]);
  if (BK_MODE == 0) {
    async_b128(ldsB0, Bm + bOff0);
    async_b128(ldsB1, Bm + bOff1);
  } else {
    const u16* gp = Bm + (long long)kk * ldb + bN + ng * 16;
    uint4 v0 = *(const uint4*)gp;
    uint4 v1 = *(const uint4*)(gp + 8);
    const u16* e0 = (const u16*)&v0;
    const u16* e1 = (const u16*)&v1;
    u16* bs = &Bs[0][(ng * 16) * 40 + kk];
#pragma unroll
    for (int j = 0; j < 8; ++j) bs[j * 40] = e0[j];
#pragma unroll
    for (int j = 0; j < 8; ++j) bs[(8 + j) * 40] = e1[j];
  }
  wait_async0();
  __syncthreads();

  // ---- main loop: prefetch tile k+1 (async -> LDS) behind tile k's WMMAs.
  int cur = 0;
  for (int k = 0; k < nsteps - 1; ++k) {
    const int kt = (k + 1) << 5;
    const unsigned boA = (cur ^ 1) ? BUFA : 0u;
    const unsigned boB = (cur ^ 1) ? BUFB : 0u;
    uint4 v0 = {}, v1 = {};
#pragma unroll
    for (int c = 0; c < 4; ++c) async_b128(ldsA[c] + boA, A + aOff[c] + kt);
    if (BK_MODE == 0) {
      async_b128(ldsB0 + boB, Bm + bOff0 + kt);
      async_b128(ldsB1 + boB, Bm + bOff1 + kt);
    } else {
      const u16* gp = Bm + (long long)(kt + kk) * ldb + bN + ng * 16;
      v0 = *(const uint4*)gp;       // prefetch into regs; stored after WMMAs
      v1 = *(const uint4*)(gp + 8);
    }

    do_tile(As[cur], Bs[cur], waveM, waveN, lane, acc);

    if (BK_MODE == 1) {
      const u16* e0 = (const u16*)&v0;
      const u16* e1 = (const u16*)&v1;
      u16* bs = &Bs[cur ^ 1][(ng * 16) * 40 + kk];
#pragma unroll
      for (int j = 0; j < 8; ++j) bs[j * 40] = e0[j];
#pragma unroll
      for (int j = 0; j < 8; ++j) bs[(8 + j) * 40] = e1[j];
    }
    wait_async0();
    __syncthreads();
    cur ^= 1;
  }
  do_tile(As[cur], Bs[cur], waveM, waveN, lane, acc);

  // Epilogue. C/D layout (ISA 7.12.2): lane = N + 16*(M>=8), vgpr r -> M = r (+8).
  const int nl = lane & 15;
  const int mh = (lane >> 4) & 1;
  u16*   Du = (u16*)Dp   + (long long)bz * sDb;
  float* Df = (float*)Dp + (long long)bz * sDb;
#pragma unroll
  for (int tm = 0; tm < 4; ++tm) {
#pragma unroll
    for (int tn = 0; tn < 4; ++tn) {
      int gm0 = bM + waveM * 64 + tm * 16 + mh * 8;
      int gn  = bN + waveN * 64 + tn * 16 + nl;
#pragma unroll
      for (int r = 0; r < 8; ++r) {
        int gm = gm0 + r;
        float v = acc[tm][tn][r];
        long long idx = (long long)gm * ldd + gn;
        if (EPI == 0)      Du[idx] = f2bf(v + auxp[gn]);
        else if (EPI == 1) Df[idx] = v * scale;
        else if (EPI == 2) Du[idx] = f2bf(v + auxp[idx]);
        else               Df[idx] = v + auxp[gn];
      }
    }
  }
}

// ------------------------------ row softmax --------------------------------
// One 256-thread workgroup per row of S[b, q, 0..2047]; adds mask[b, k],
// writes bf16 probabilities.
__global__ __launch_bounds__(256) void softmax_rows(
    const float* __restrict__ S, const float* __restrict__ mask,
    u16* __restrict__ P, int cols, int rows_per_batch)
{
  __shared__ float red[256];
  const int row = blockIdx.x;
  const int b   = row / rows_per_batch;
  const int t   = threadIdx.x;
  const float* s  = S    + (long long)row * cols;
  const float* mk = mask + (long long)b * cols;

  float vals[8];
  float mx = -3.4e38f;
#pragma unroll
  for (int i = 0; i < 8; ++i) {
    int k = t + i * 256;
    float v = s[k] + mk[k];
    vals[i] = v;
    mx = fmaxf(mx, v);
  }
  red[t] = mx; __syncthreads();
  for (int w = 128; w > 0; w >>= 1) { if (t < w) red[t] = fmaxf(red[t], red[t + w]); __syncthreads(); }
  mx = red[0]; __syncthreads();

  float sum = 0.f;
#pragma unroll
  for (int i = 0; i < 8; ++i) { vals[i] = __expf(vals[i] - mx); sum += vals[i]; }
  red[t] = sum; __syncthreads();
  for (int w = 128; w > 0; w >>= 1) { if (t < w) red[t] += red[t + w]; __syncthreads(); }
  float inv = 1.0f / red[0];

  u16* p = P + (long long)row * cols;
#pragma unroll
  for (int i = 0; i < 8; ++i) p[t + i * 256] = f2bf(vals[i] * inv);
}

// ------------------------------- launcher ----------------------------------
extern "C" void kernel_launch(void* const* d_in, const int* in_sizes, int n_in,
                              void* d_out, int out_size, void* d_ws, size_t ws_size,
                              hipStream_t stream) {
  (void)in_sizes; (void)n_in; (void)out_size; (void)ws_size;
  const float* hs   = (const float*)d_in[0];   // [B,1,SQ,HID]
  const float* chs  = (const float*)d_in[1];   // [B,1,SK,HID]
  const float* mask = (const float*)d_in[2];   // [B,1,1,SK]
  const float* Wq = (const float*)d_in[3];  const float* bq = (const float*)d_in[4];
  const float* Wk = (const float*)d_in[5];  const float* bk = (const float*)d_in[6];
  const float* Wv = (const float*)d_in[7];  const float* bv = (const float*)d_in[8];
  const float* Wd = (const float*)d_in[9];  const float* bd = (const float*)d_in[10];
  float* out = (float*)d_out;

  const long long BSQ  = (long long)NB * SEQ;   // 16384
  const long long NELX = BSQ * HID;             // 33.5M
  const long long NELW = (long long)HID * HID;  // 4.2M

  char* w = (char*)d_ws;
  size_t off = 0;
  auto carve = [&](size_t bytes) {
    char* p = w + off;
    off = (off + bytes + 255) & ~(size_t)255;
    return p;
  };
  u16*   Xb  = (u16*)carve(NELX * 2);                    // bf16 hidden_states
  u16*   XCb = (u16*)carve(NELX * 2);                    // bf16 cross_hidden
  u16*   Wqb = (u16*)carve(NELW * 2);
  u16*   Wkb = (u16*)carve(NELW * 2);
  u16*   Wvb = (u16*)carve(NELW * 2);
  u16*   Wdb = (u16*)carve(NELW * 2);
  u16*   Qb  = (u16*)carve(NELX * 2);
  u16*   Kb  = (u16*)carve(NELX * 2);
  u16*   Vb  = (u16*)carve(NELX * 2);
  float* S   = (float*)carve((size_t)NB * SEQ * SEQ * 4); // f32 scores (128 MiB)
  u16*   Pb  = Xb;   // reuse: X-bf16 dead after Q projection
  u16*   Cb  = XCb;  // reuse: XC-bf16 dead after V projection

  dim3 blk(256);
  // fp32 -> bf16 conversions
  cvt_f32_bf16<<<dim3(2048), blk, 0, stream>>>(hs,  Xb,  NELX);
  cvt_f32_bf16<<<dim3(2048), blk, 0, stream>>>(chs, XCb, NELX);
  cvt_f32_bf16<<<dim3(512),  blk, 0, stream>>>(Wq,  Wqb, NELW);
  cvt_f32_bf16<<<dim3(512),  blk, 0, stream>>>(Wk,  Wkb, NELW);
  cvt_f32_bf16<<<dim3(512),  blk, 0, stream>>>(Wv,  Wvb, NELW);
  cvt_f32_bf16<<<dim3(512),  blk, 0, stream>>>(Wd,  Wdb, NELW);

  const float scale = 1.0f / sqrtf((float)HID);

  // Q/K/V projections: [16384,2048] = X[16384,2048] * W[2048,2048]^T + b
  dim3 gproj(HID / 128, (int)(BSQ / 256), 1);   // (16, 64)
  gemm_wmma<0, 0><<<gproj, blk, 0, stream>>>(Xb,  0, Wqb, 0, Qb, 0, bq, 0,
      (int)BSQ, HID, HID, HID, HID, HID, 1.0f);
  gemm_wmma<0, 0><<<gproj, blk, 0, stream>>>(XCb, 0, Wkb, 0, Kb, 0, bk, 0,
      (int)BSQ, HID, HID, HID, HID, HID, 1.0f);
  gemm_wmma<0, 0><<<gproj, blk, 0, stream>>>(XCb, 0, Wvb, 0, Vb, 0, bv, 0,
      (int)BSQ, HID, HID, HID, HID, HID, 1.0f);

  // Scores: per batch  S = scale * Q[2048,2048] * K[2048,2048]^T  (f32 out)
  dim3 gatt(SEQ / 128, SEQ / 256, NB);          // (16, 8, 8)
  gemm_wmma<0, 1><<<gatt, blk, 0, stream>>>(
      Qb, (long long)SEQ * HID, Kb, (long long)SEQ * HID,
      S,  (long long)SEQ * SEQ, mask, 0,
      SEQ, SEQ, HID, HID, HID, SEQ, scale);

  // Softmax rows (adds mask), bf16 probabilities
  softmax_rows<<<dim3(NB * SEQ), blk, 0, stream>>>(S, mask, Pb, SEQ, SEQ);

  // Context + residual: Cb = bf16( P[2048,2048] * V[2048,2048] + hidden_f32 )
  gemm_wmma<1, 2><<<dim3(HID / 128, SEQ / 256, NB), blk, 0, stream>>>(
      Pb, (long long)SEQ * SEQ, Vb, (long long)SEQ * HID,
      Cb, (long long)SEQ * HID, hs, (long long)SEQ * HID,
      SEQ, HID, SEQ, SEQ, HID, HID, 1.0f);

  // Output projection: out_f32 = Cb[16384,2048] * Wd^T + bd
  gemm_wmma<0, 3><<<gproj, blk, 0, stream>>>(Cb, 0, Wdb, 0, out, 0, bd, 0,
      (int)BSQ, HID, HID, HID, HID, HID, 1.0f);
}
